// MultiHeadAttention_83030307766542
// MI455X (gfx1250) — compile-verified, matrix operand order checked by reading, not run on hardware
//
#include <hip/hip_runtime.h>
#include <hip/hip_bf16.h>

#define EMBED 1024
#define HEADS 16
#define HEAD_DIM 64
#define SEQ 2048
#define BATCH 2

typedef __attribute__((ext_vector_type(16))) __bf16 v16bf;
typedef __attribute__((ext_vector_type(8)))  __bf16 v8bf;
typedef __attribute__((ext_vector_type(8)))  float  v8f;
typedef __attribute__((ext_vector_type(4)))  unsigned int u32x4;
typedef __attribute__((ext_vector_type(8)))  int i32x8;
typedef __attribute__((ext_vector_type(4)))  int i32x4;

// CDNA5 16-bit A/B fragment (16x32): lane-half 0 holds K 0..7 / 16..23,
// lane-half 1 holds K 8..15 / 24..31 -> two 16-byte loads per lane.
__device__ __forceinline__ v16bf frag_from_ptr(const __bf16* rowp, int half) {
  const v8bf lo = *(const v8bf*)(rowp + half * 8);
  const v8bf hi = *(const v8bf*)(rowp + 16 + half * 8);
  v16bf f;
#pragma unroll
  for (int e = 0; e < 8; ++e) {
    f[e] = lo[e];
    f[8 + e] = hi[e];
  }
  return f;
}

// LDS byte offset of a __shared__ object (flat LDS-aperture addr[31:0] = offset)
__device__ __forceinline__ unsigned lds_off(const void* p) {
  return (unsigned)(uintptr_t)p;
}

// ---------------------------------------------------------------------------
// Tensor Data Mover: 2D tile load (global -> LDS), bf16 elements.
//   tile_d0: contiguous elements per row; tile_d1: rows
//   stride0: global row stride in elements
//   pad_int/pad_amt: D# pad codes. Row of tile_d0 bf16 = tile_d0/2 dwords:
//     32 dwords -> pad_int=4, 16 dwords -> pad_int=3; pad 4 dwords -> pad_amt=3.
// Issue from one wave; descriptors must be wave-uniform.
// (6-arg builtin form on this toolchain: g0, g1, g2, g3, g4, cpol.)
// ---------------------------------------------------------------------------
__device__ __forceinline__ void tdm_load_2d_bf16(
    unsigned lds_addr, const void* gptr,
    unsigned tile_d0, unsigned tile_d1, unsigned long stride0,
    unsigned pad_int, unsigned pad_amt) {
  const unsigned long ga = (unsigned long)(uintptr_t)gptr;
  u32x4 g0;
  g0[0] = 1u;                                   // count=1, user mode
  g0[1] = lds_addr;                             // lds_addr [63:32]
  g0[2] = (unsigned)(ga & 0xFFFFFFFFu);         // global_addr [95:64]
  g0[3] = (unsigned)((ga >> 32) & 0x01FFFFFFu)  // global_addr [120:96]
          | (2u << 30);                         // type=2 ("image") [127:126]

  const unsigned td0 = (unsigned)stride0;       // tensor_dim0 (OOB never hit)
  const unsigned td1 = 0x7FFFFFFFu;             // tensor_dim1: huge
  i32x8 g1;
  g1[0] = (int)((1u << 16)                      // data_size = 2B
                | (1u << 20)                    // pad_enable
                | (pad_int << 22) | (pad_amt << 25));
  g1[1] = (int)((td0 & 0xFFFFu) << 16);         // tensor_dim0[15:0]
  g1[2] = (int)((td0 >> 16) | ((td1 & 0xFFFFu) << 16));
  g1[3] = (int)((td1 >> 16) | (tile_d0 << 16)); // tile_dim0
  g1[4] = (int)(tile_d1 & 0xFFFFu);             // tile_dim1, tile_dim2=0
  g1[5] = (int)(unsigned)(stride0 & 0xFFFFFFFFu);      // dim0_stride lo
  g1[6] = (int)((unsigned)(stride0 >> 32) & 0xFFFFu);  // dim0_stride hi
  g1[7] = 0;                                    // dim1_stride (unused, 2D)
  const i32x4 z4 = {0, 0, 0, 0};
  const i32x8 z8i = {0, 0, 0, 0, 0, 0, 0, 0};
  __builtin_amdgcn_tensor_load_to_lds(g0, g1, z4, z4, z8i, 0);
}

// ---------------------------------------------------------------------------
// Weight pre-pass: W[K][N] fp32 -> Wt[N][K] bf16 (transpose + convert).
// ---------------------------------------------------------------------------
__global__ __launch_bounds__(256) void transpose_bf16_kernel(
    const float* __restrict__ W, __bf16* __restrict__ Wt, int N, int K) {
  __shared__ float tile[32][33];
  const int n0 = blockIdx.x * 32;
  const int k0 = blockIdx.y * 32;
  const int tx = threadIdx.x & 31;
  const int ty = threadIdx.x >> 5;
#pragma unroll
  for (int i = 0; i < 32; i += 8)
    tile[ty + i][tx] = W[(long)(k0 + ty + i) * N + n0 + tx];
  __syncthreads();
#pragma unroll
  for (int i = 0; i < 32; i += 8)
    Wt[(long)(n0 + ty + i) * K + k0 + tx] = (__bf16)tile[tx][ty + i];
}

// ---------------------------------------------------------------------------
// GEMM: out[M,N] = A[M,K] @ Wt^T + bias (Wt is [N][K] bf16).
// Block tile 128x64, double-buffered LDS; W tile (and A tile when A is bf16)
// staged by the Tensor Data Mover; fp32 A staged cooperatively with convert.
// OUT_LAYOUT: 0 = bf16 row-major, 1 = bf16 head-transposed [B][H][Dh][S],
//             2 = f32 row-major.
// ---------------------------------------------------------------------------
template <typename AT, int OUT_LAYOUT>
__global__ __launch_bounds__(256) void wmma_gemm_kernel(
    const AT* __restrict__ A, const __bf16* __restrict__ Wt,
    const float* __restrict__ bias, void* __restrict__ outp,
    int M, int N, int K) {
  __shared__ __bf16 As[2][128][40];  // 32 elems + 8 pad (16B aligned rows)
  __shared__ __bf16 Ws[2][64][40];

  const int tid = threadIdx.x;
  const int lane = tid & 31;
  const int wave = tid >> 5;
  const int half = (lane >> 4) & 1;
  const int ml = lane & 15;
  const int rowBlock = blockIdx.x * 128;
  const int col0 = blockIdx.y * 64;

  const v8f z8 = {0.f, 0.f, 0.f, 0.f, 0.f, 0.f, 0.f, 0.f};
  v8f acc[4] = {z8, z8, z8, z8};

  const int ar = tid >> 1;        // fp32 A stage: row 0..127
  const int ak = (tid & 1) * 16;  //               k-offset 0/16

  // ---- prologue: stage k0 = 0 into buffer 0 ----
  if (wave == 0) {
    tdm_load_2d_bf16(lds_off(&Ws[0][0][0]), Wt + (long)col0 * K,
                     32, 64, (unsigned long)K, 3, 3);
    if constexpr (sizeof(AT) == 2) {
      tdm_load_2d_bf16(lds_off(&As[0][0][0]), A + (long)rowBlock * K,
                       32, 128, (unsigned long)K, 3, 3);
    }
  }
  if constexpr (sizeof(AT) == 4) {
    const float4* s4 = (const float4*)(A + (long)(rowBlock + ar) * K + ak);
#pragma unroll
    for (int q4 = 0; q4 < 4; ++q4) {
      const float4 f = s4[q4];
      __bf16* dst = &As[0][ar][ak + q4 * 4];
      dst[0] = (__bf16)f.x; dst[1] = (__bf16)f.y;
      dst[2] = (__bf16)f.z; dst[3] = (__bf16)f.w;
    }
  }
  if (wave == 0) __builtin_amdgcn_s_wait_tensorcnt(0);
  __syncthreads();

  for (int k0 = 0; k0 < K; k0 += 32) {
    const int cur = (k0 >> 5) & 1;
    const int nxt = cur ^ 1;
    const bool have_next = (k0 + 32 < K);

    // ---- stage next tile into the other buffer (overlaps compute) ----
    if (wave == 0 && have_next) {
      tdm_load_2d_bf16(lds_off(&Ws[nxt][0][0]),
                       Wt + (long)col0 * K + k0 + 32,
                       32, 64, (unsigned long)K, 3, 3);
      if constexpr (sizeof(AT) == 2) {
        tdm_load_2d_bf16(lds_off(&As[nxt][0][0]),
                         A + (long)rowBlock * K + k0 + 32,
                         32, 128, (unsigned long)K, 3, 3);
      }
    }
    if constexpr (sizeof(AT) == 4) {
      if (have_next) {
        const float4* s4 =
            (const float4*)(A + (long)(rowBlock + ar) * K + k0 + 32 + ak);
#pragma unroll
        for (int q4 = 0; q4 < 4; ++q4) {
          const float4 f = s4[q4];
          __bf16* dst = &As[nxt][ar][ak + q4 * 4];
          dst[0] = (__bf16)f.x; dst[1] = (__bf16)f.y;
          dst[2] = (__bf16)f.z; dst[3] = (__bf16)f.w;
        }
      }
    }

    // ---- compute on current buffer: batch all fragment loads first ----
    const v16bf a = frag_from_ptr(&As[cur][wave * 16 + ml][0], half);
    v16bf bfr[4];
#pragma unroll
    for (int n = 0; n < 4; ++n)
      bfr[n] = frag_from_ptr(&Ws[cur][n * 16 + ml][0], half);
#pragma unroll
    for (int n = 0; n < 4; ++n) {
      acc[n] = __builtin_amdgcn_wmma_f32_16x16x32_bf16(
          false, a, false, bfr[n], (short)0, acc[n], false, false);
    }

    if (wave == 0) __builtin_amdgcn_s_wait_tensorcnt(0);
    __syncthreads();
  }

  // C/D layout: element (m = r + 8*half, n = ml)
#pragma unroll
  for (int n = 0; n < 4; ++n) {
    const int col = col0 + n * 16 + ml;
    const float bias_v = bias[col];
#pragma unroll
    for (int r = 0; r < 8; ++r) {
      const long row = rowBlock + wave * 16 + r + half * 8;
      const float val = acc[n][r] + bias_v;
      if constexpr (OUT_LAYOUT == 0) {
        ((__bf16*)outp)[row * N + col] = (__bf16)val;
      } else if constexpr (OUT_LAYOUT == 1) {
        const long b = row / SEQ, s = row % SEQ;
        const int h = col >> 6, d = col & 63;
        ((__bf16*)outp)[(((long)b * HEADS + h) * HEAD_DIM + d) * SEQ + s] =
            (__bf16)val;
      } else {
        ((float*)outp)[row * N + col] = val;
      }
    }
  }
}

// ---------------------------------------------------------------------------
// Flash attention. Block = 8 waves (one 16-query tile each) sharing (b,h).
// K(32x64) and V^T(64x32) tiles are DMA'd into double-buffered LDS by the
// Tensor Data Mover (wave 0), overlapping compute. S^T = K @ Q^T keeps the
// softmax key-reduction in-lane, and exp(S^T) is directly the B-fragment of
// O^T += V^T @ P^T. Mask bias comes from 4 float4 loads (no bpermute).
// ---------------------------------------------------------------------------
__global__ __launch_bounds__(256) void flash_attn_kernel(
    const __bf16* __restrict__ Qp, const __bf16* __restrict__ Kp,
    const __bf16* __restrict__ Vt, const float* __restrict__ mask,
    __bf16* __restrict__ ctx) {
  __shared__ __bf16 Ks[2][32][72];  // 32 keys x 64 d (+8 pad)
  __shared__ __bf16 Vs[2][64][40];  // 64 d x 32 keys (+8 pad)

  const int tid = threadIdx.x;
  const int lane = tid & 31;
  const int wave = tid >> 5;
  const int half = (lane >> 4) & 1;
  const int ml = lane & 15;
  const int b = blockIdx.z;
  const int h = blockIdx.y;
  const int q0 = (blockIdx.x * 8 + wave) * 16;

  const __bf16* kbase = Kp + (long)(b * SEQ) * EMBED + h * HEAD_DIM;
  const __bf16* vbase = Vt + (long)(b * HEADS + h) * HEAD_DIM * SEQ;
  const float* mrow0 = mask + (long)b * SEQ;

  // ---- prologue: DMA tile kb = 0 into buffer 0 ----
  if (wave == 0) {
    tdm_load_2d_bf16(lds_off(&Ks[0][0][0]), kbase, 64, 32, EMBED, 4, 3);
    tdm_load_2d_bf16(lds_off(&Vs[0][0][0]), vbase, 32, 64, SEQ, 3, 3);
  }

  // Q^T as B-fragments (k = d, n = query), loaded once, b128 per lane
  const __bf16* qbase = Qp + (long)(b * SEQ + q0 + ml) * EMBED + h * HEAD_DIM;
  v16bf qt[2];
  qt[0] = frag_from_ptr(qbase + 0, half);
  qt[1] = frag_from_ptr(qbase + 32, half);

  const v8f z8 = {0.f, 0.f, 0.f, 0.f, 0.f, 0.f, 0.f, 0.f};
  v8f o[4] = {z8, z8, z8, z8};  // O^T: 64d x 16q
  float m_run = -INFINITY;
  float l_run = 0.f;
  const float scale = 0.125f;  // 1/sqrt(HEAD_DIM)

  if (wave == 0) __builtin_amdgcn_s_wait_tensorcnt(0);
  __syncthreads();

  for (int kb = 0; kb < SEQ; kb += 32) {
    const int cur = (kb >> 5) & 1;
    const int nxt = cur ^ 1;

    // ---- DMA next tile (overlaps compute) ----
    if (wave == 0 && kb + 32 < SEQ) {
      tdm_load_2d_bf16(lds_off(&Ks[nxt][0][0]),
                       kbase + (long)(kb + 32) * EMBED, 64, 32, EMBED, 4, 3);
      tdm_load_2d_bf16(lds_off(&Vs[nxt][0][0]),
                       vbase + (kb + 32), 32, 64, SEQ, 3, 3);
    }

    // ---- batch all K fragment loads, then 4 WMMAs: S^T = K @ Q^T ----
    v16bf kf[2][2];
#pragma unroll
    for (int kh = 0; kh < 2; ++kh) {
      kf[kh][0] = frag_from_ptr(&Ks[cur][kh * 16 + ml][0], half);
      kf[kh][1] = frag_from_ptr(&Ks[cur][kh * 16 + ml][32], half);
    }
    // batch V^T fragment loads too (independent of S^T computation)
    v16bf vf[4];
#pragma unroll
    for (int c = 0; c < 4; ++c)
      vf[c] = frag_from_ptr(&Vs[cur][c * 16 + ml][0], half);

    v8f s[2] = {z8, z8};
#pragma unroll
    for (int kh = 0; kh < 2; ++kh) {
      s[kh] = __builtin_amdgcn_wmma_f32_16x16x32_bf16(
          false, kf[kh][0], false, qt[0], (short)0, s[kh], false, false);
      s[kh] = __builtin_amdgcn_wmma_f32_16x16x32_bf16(
          false, kf[kh][1], false, qt[1], (short)0, s[kh], false, false);
    }

    // ---- mask bias: 4x float4, keys {half*8..} and {16+half*8..} ----
    float mbias[16];
    {
      const float* mr = mrow0 + kb;
      const float4 a0 = *(const float4*)(mr + half * 8);
      const float4 a1 = *(const float4*)(mr + half * 8 + 4);
      const float4 a2 = *(const float4*)(mr + 16 + half * 8);
      const float4 a3 = *(const float4*)(mr + 16 + half * 8 + 4);
      mbias[0] = a0.x; mbias[1] = a0.y; mbias[2] = a0.z; mbias[3] = a0.w;
      mbias[4] = a1.x; mbias[5] = a1.y; mbias[6] = a1.z; mbias[7] = a1.w;
      mbias[8] = a2.x; mbias[9] = a2.y; mbias[10] = a2.z; mbias[11] = a2.w;
      mbias[12] = a3.x; mbias[13] = a3.y; mbias[14] = a3.z; mbias[15] = a3.w;
    }

    // ---- online softmax over keys ----
    float pvals[16];
    float lmax = -INFINITY;
#pragma unroll
    for (int kh = 0; kh < 2; ++kh) {
#pragma unroll
      for (int r = 0; r < 8; ++r) {
        const float sv = s[kh][r] * scale + mbias[kh * 8 + r] * (-1e9f);
        pvals[kh * 8 + r] = sv;
        lmax = fmaxf(lmax, sv);
      }
    }
    lmax = fmaxf(lmax, __shfl_xor(lmax, 16, 32));

    const float m_new = fmaxf(m_run, lmax);
    const float alpha = __expf(m_run - m_new);
    float lsum = 0.f;
#pragma unroll
    for (int i = 0; i < 16; ++i) {
      const float p = __expf(pvals[i] - m_new);
      pvals[i] = p;
      lsum += p;
    }
    lsum += __shfl_xor(lsum, 16, 32);
    l_run = l_run * alpha + lsum;
    m_run = m_new;

#pragma unroll
    for (int c = 0; c < 4; ++c)
#pragma unroll
      for (int r = 0; r < 8; ++r) o[c][r] *= alpha;

    // P^T is already in B-fragment element order
    v16bf pf;
#pragma unroll
    for (int e = 0; e < 16; ++e) pf[e] = (__bf16)pvals[e];

    // ---- O^T += V^T @ P^T : 4 WMMAs (fragments already loaded) ----
#pragma unroll
    for (int c = 0; c < 4; ++c) {
      o[c] = __builtin_amdgcn_wmma_f32_16x16x32_bf16(
          false, vf[c], false, pf, (short)0, o[c], false, false);
    }

    if (wave == 0) __builtin_amdgcn_s_wait_tensorcnt(0);
    __syncthreads();
  }

  // epilogue: ctx[b*S+q][h*64+d], contiguous over r -> b128 stores
  const float linv = (l_run > 0.f) ? (1.f / l_run) : 0.f;
  __bf16* cbase = ctx + (long)(b * SEQ + q0 + ml) * EMBED + h * HEAD_DIM;
#pragma unroll
  for (int c = 0; c < 4; ++c) {
    v8bf ov;
#pragma unroll
    for (int r = 0; r < 8; ++r) ov[r] = (__bf16)(o[c][r] * linv);
    *(v8bf*)(cbase + c * 16 + half * 8) = ov;
  }
}

extern "C" void kernel_launch(void* const* d_in, const int* in_sizes, int n_in,
                              void* d_out, int out_size, void* d_ws, size_t ws_size,
                              hipStream_t stream) {
  const float* v    = (const float*)d_in[0];
  const float* k    = (const float*)d_in[1];
  const float* q    = (const float*)d_in[2];
  const float* mask = (const float*)d_in[3];
  const float* wq = (const float*)d_in[4];
  const float* bq = (const float*)d_in[5];
  const float* wk = (const float*)d_in[6];
  const float* bk = (const float*)d_in[7];
  const float* wv = (const float*)d_in[8];
  const float* bv = (const float*)d_in[9];
  const float* wo = (const float*)d_in[10];
  const float* bo = (const float*)d_in[11];

  const int M = BATCH * SEQ;  // 4096
  const size_t actBytes = (size_t)M * EMBED * sizeof(__bf16);      // 8 MB
  const size_t wBytes   = (size_t)EMBED * EMBED * sizeof(__bf16);  // 2 MB
  char* ws = (char*)d_ws;
  __bf16* Qp  = (__bf16*)(ws);
  __bf16* Kp  = (__bf16*)(ws + 1 * actBytes);
  __bf16* Vth = (__bf16*)(ws + 2 * actBytes);  // [B][H][Dh][S]
  __bf16* ctx = (__bf16*)(ws + 3 * actBytes);
  __bf16* Wqt = (__bf16*)(ws + 4 * actBytes);
  __bf16* Wkt = (__bf16*)(ws + 4 * actBytes + 1 * wBytes);
  __bf16* Wvt = (__bf16*)(ws + 4 * actBytes + 2 * wBytes);
  __bf16* Wot = (__bf16*)(ws + 4 * actBytes + 3 * wBytes);

  dim3 blk(256);
  dim3 tgrid(EMBED / 32, EMBED / 32);
  hipLaunchKernelGGL(transpose_bf16_kernel, tgrid, blk, 0, stream, wq, Wqt, EMBED, EMBED);
  hipLaunchKernelGGL(transpose_bf16_kernel, tgrid, blk, 0, stream, wk, Wkt, EMBED, EMBED);
  hipLaunchKernelGGL(transpose_bf16_kernel, tgrid, blk, 0, stream, wv, Wvt, EMBED, EMBED);
  hipLaunchKernelGGL(transpose_bf16_kernel, tgrid, blk, 0, stream, wo, Wot, EMBED, EMBED);

  dim3 ggrid(M / 128, EMBED / 64);  // (32, 16)
  hipLaunchKernelGGL((wmma_gemm_kernel<float, 0>), ggrid, blk, 0, stream,
                     q, Wqt, bq, (void*)Qp, M, EMBED, EMBED);
  hipLaunchKernelGGL((wmma_gemm_kernel<float, 0>), ggrid, blk, 0, stream,
                     k, Wkt, bk, (void*)Kp, M, EMBED, EMBED);
  hipLaunchKernelGGL((wmma_gemm_kernel<float, 1>), ggrid, blk, 0, stream,
                     v, Wvt, bv, (void*)Vth, M, EMBED, EMBED);

  dim3 fgrid(SEQ / 128, HEADS, BATCH);  // (16, 16, 2)
  hipLaunchKernelGGL(flash_attn_kernel, fgrid, blk, 0, stream,
                     Qp, Kp, Vth, mask, ctx);

  hipLaunchKernelGGL((wmma_gemm_kernel<__bf16, 2>), ggrid, blk, 0, stream,
                     ctx, Wot, bo, d_out, M, EMBED, EMBED);
}
